// David_46523085750370
// MI455X (gfx1250) — matrix operand order repaired and checked by reading, not compiled, not run on hardware
//
#include <hip/hip_runtime.h>
#include <hip/hip_bf16.h>
#include <math.h>

// ---------------------------------------------------------------------------
// Types
// ---------------------------------------------------------------------------
typedef __attribute__((ext_vector_type(16))) __bf16 v16bf;
typedef __attribute__((ext_vector_type(8)))  float  v8f;
typedef __attribute__((ext_vector_type(4)))  float  vf4;
typedef __attribute__((ext_vector_type(4)))  __bf16 v4bf;

#define BM 128
#define BN 128
#define BK 32
#define LDSK 40   // padded LDS row stride in bf16 elems (80B -> conflict-free b128)

struct B32x2 { uint4 a; uint4 b; };

__device__ __forceinline__ v16bf ld_frag32(const unsigned short* p0,
                                           const unsigned short* p1) {
  B32x2 u;
  u.a = *(const uint4*)p0;
  u.b = *(const uint4*)p1;
  return __builtin_bit_cast(v16bf, u);
}

// ---------------------------------------------------------------------------
// gfx1250 async global->LDS copy (16B per lane, no data VGPRs, ASYNCcnt)
// VDST = LDS byte address (low 32 bits of generic pointer == LDS offset),
// VADDR = 64-bit global address.  ISA 15.18.3 op 98.
// ---------------------------------------------------------------------------
__device__ __forceinline__ void async_b128(const void* g, void* lds) {
  asm volatile("global_load_async_to_lds_b128 %0, %1, off"
               :
               : "v"((unsigned)(unsigned long long)lds),
                 "v"((unsigned long long)g)
               : "memory");
}
__device__ __forceinline__ void wait_async0() {
  asm volatile("s_wait_asynccnt 0x0" ::: "memory");
}

// ---------------------------------------------------------------------------
// Bulk f32 -> bf16 conversion (v_cvt_pk_bf16_f32 via convertvector)
// ---------------------------------------------------------------------------
__global__ __launch_bounds__(256) void cvt_f32_to_bf16_kernel(
    const float* __restrict__ src, unsigned short* __restrict__ dst, long n4)
{
  const long i = (long)blockIdx.x * 256 + threadIdx.x;
  if (i >= n4) return;
  vf4 v = *(const vf4*)(src + i * 4);
  v4bf b = __builtin_convertvector(v, v4bf);
  *(unsigned long long*)(dst + i * 4) = __builtin_bit_cast(unsigned long long, b);
}

// ---------------------------------------------------------------------------
// GEMM: C[M,N] = A[M,K] * W[N,K]^T + bias1[N] (+ bias2[N])
// A, W bf16 row-major (row stride == Kd). Output f32 or bf16.
// Async global->LDS double-buffered pipeline, 1 barrier per 32-K slab.
// ---------------------------------------------------------------------------
template <bool OUT_BF16, bool HAS_B2>
__global__ __launch_bounds__(256) void gemm_nt_bf16_wmma(
    const unsigned short* __restrict__ A, const unsigned short* __restrict__ W,
    const float* __restrict__ bias1, const float* __restrict__ bias2,
    void* __restrict__ Cout, int Kd, int ldc)
{
  __shared__ unsigned short sA[2][BM * LDSK];
  __shared__ unsigned short sB[2][BN * LDSK];

  const int tid  = threadIdx.x;
  const int lane = tid & 31;
  const int wave = tid >> 5;
  const int wm   = wave & 3;      // 4 waves along M -> 32 rows each
  const int wn   = wave >> 2;     // 2 waves along N -> 64 cols each
  const int half = lane >> 4;
  const int r16  = lane & 15;

  const long m0 = (long)blockIdx.y * BM;
  const long n0 = (long)blockIdx.x * BN;

  // staging map: row = 32 bf16 = 4 chunks of 8; 64 rows per pass, 2 passes
  const int srow = tid >> 2;          // 0..63
  const int scol = (tid & 3) * 8;     // 0,8,16,24 (bf16 elems)

  const unsigned short* pa = A + (m0 + srow) * (long)Kd + scol;
  const unsigned short* pw = W + (n0 + srow) * (long)Kd + scol;
  const long rstep = 64L * (long)Kd;  // 64 rows ahead
  const int  lofs  = srow * LDSK + scol;

  const v8f vz = {0.f,0.f,0.f,0.f,0.f,0.f,0.f,0.f};
  v8f acc[2][4];
  for (int i = 0; i < 2; ++i)
    for (int j = 0; j < 4; ++j) acc[i][j] = vz;

  // prologue: async-stage first K-slab into buffer 0
  async_b128(pa,         &sA[0][lofs]);
  async_b128(pa + rstep, &sA[0][lofs + 64 * LDSK]);
  async_b128(pw,         &sB[0][lofs]);
  async_b128(pw + rstep, &sB[0][lofs + 64 * LDSK]);
  wait_async0();

  int it = 0;
  for (int k0 = 0; k0 < Kd; k0 += BK, ++it) {
    const int cur = it & 1;
    __syncthreads();   // buf[cur] staged & visible; prior readers of buf[cur^1] done

    const bool has_next = (k0 + BK) < Kd;
    if (has_next) {
      const int nxt = cur ^ 1;
      const long ko = k0 + BK;
      async_b128(pa + ko,         &sA[nxt][lofs]);
      async_b128(pa + ko + rstep, &sA[nxt][lofs + 64 * LDSK]);
      async_b128(pw + ko,         &sB[nxt][lofs]);
      async_b128(pw + ko + rstep, &sB[nxt][lofs + 64 * LDSK]);
    }

    // ---- fragment loads (ISA 7.12.2 layouts) ----
    v16bf af[2], bfg[4];
#pragma unroll
    for (int i = 0; i < 2; ++i) {
      // A 16x32 bf16: lanes 0-15 row M=r16 hold K 0..7 & 16..23; lanes 16-31 K 8..15 & 24..31
      const unsigned short* p = &sA[cur][(wm * 32 + i * 16 + r16) * LDSK + half * 8];
      af[i] = ld_frag32(p, p + 16);
    }
#pragma unroll
    for (int j = 0; j < 4; ++j) {
      // B 32x16 bf16: lanes 0-15 col N=r16 hold K 0..7 & 16..23; lanes 16-31 K 8..15 & 24..31
      const unsigned short* p = &sB[cur][(wn * 64 + j * 16 + r16) * LDSK + half * 8];
      bfg[j] = ld_frag32(p, p + 16);
    }

#pragma unroll
    for (int i = 0; i < 2; ++i)
#pragma unroll
      for (int j = 0; j < 4; ++j)
        acc[i][j] = __builtin_amdgcn_wmma_f32_16x16x32_bf16(
            false, af[i], false, bfg[j], (short)0, acc[i][j], false, false);

    if (has_next) wait_async0();   // own async copies done before next barrier
  }

  // ---- epilogue: C/D layout: lanes 0-15 M=e, lanes 16-31 M=8+e ----
#pragma unroll
  for (int i = 0; i < 2; ++i) {
#pragma unroll
    for (int j = 0; j < 4; ++j) {
      const long nn = n0 + wn * 64 + j * 16 + r16;
      const long mb = m0 + wm * 32 + i * 16 + half * 8;
      float bv = bias1[nn];
      if constexpr (HAS_B2) bv += bias2[nn];
      if constexpr (OUT_BF16) {
        unsigned short* C = (unsigned short*)Cout;
#pragma unroll
        for (int e = 0; e < 8; ++e)
          C[(mb + e) * (long)ldc + nn] =
              __builtin_bit_cast(unsigned short, (__bf16)(acc[i][j][e] + bv));
      } else {
        float* C = (float*)Cout;
#pragma unroll
        for (int e = 0; e < 8; ++e)
          C[(mb + e) * (long)ldc + nn] = acc[i][j][e] + bv;
      }
    }
  }
}

// ---------------------------------------------------------------------------
// Routed attention: scores/softmax over LW=3, out-norm over D=64, mean over H=16
// ---------------------------------------------------------------------------
__global__ __launch_bounds__(256) void attn_route_kernel(
    const float* __restrict__ Q, const float* __restrict__ Km,
    const float* __restrict__ Vm, const int* __restrict__ routes,
    const float* __restrict__ temp, float* __restrict__ si)
{
  const int h  = threadIdx.x;   // 16 heads
  const int bl = threadIdx.y;   // 16 batch rows per block
  const long b = (long)blockIdx.x * 16 + bl;
  __shared__ float red[16][16][5];

  float q[64];
  const float* qp = Q + b * 1024 + h * 64;
#pragma unroll
  for (int d = 0; d < 64; ++d) q[d] = qp[d];
  const float scl = 0.125f / fabsf(temp[0]);   // 1/sqrt(64)/|T|

  for (int s = 0; s < 5; ++s) {
    const int r0 = routes[s * 3 + 0];
    const int r1 = routes[s * 3 + 1];
    const int r2 = routes[s * 3 + 2];
    const float* k0p = Km + (b * 5 + r0) * 1024 + h * 64;
    const float* k1p = Km + (b * 5 + r1) * 1024 + h * 64;
    const float* k2p = Km + (b * 5 + r2) * 1024 + h * 64;
    float s0 = 0.f, s1 = 0.f, s2 = 0.f;
#pragma unroll
    for (int d = 0; d < 64; ++d) {
      s0 += q[d] * k0p[d]; s1 += q[d] * k1p[d]; s2 += q[d] * k2p[d];
    }
    s0 *= scl; s1 *= scl; s2 *= scl;
    const float mx = fmaxf(s0, fmaxf(s1, s2));
    float e0 = expf(s0 - mx), e1 = expf(s1 - mx), e2 = expf(s2 - mx);
    const float inv = 1.f / (e0 + e1 + e2);
    e0 *= inv; e1 *= inv; e2 *= inv;
    const float* v0p = Vm + (b * 5 + r0) * 1024 + h * 64;
    const float* v1p = Vm + (b * 5 + r1) * 1024 + h * 64;
    const float* v2p = Vm + (b * 5 + r2) * 1024 + h * 64;
    float nrm = 0.f;
#pragma unroll
    for (int d = 0; d < 64; ++d) {
      const float o = e0 * v0p[d] + e1 * v1p[d] + e2 * v2p[d];
      nrm += o * o;
    }
    red[bl][h][s] = sqrtf(nrm);
  }
  __syncthreads();
  if (h < 5) {
    float acc = 0.f;
#pragma unroll
    for (int hh = 0; hh < 16; ++hh) acc += red[bl][hh][h];
    si[b * 5 + h] = acc * (1.f / 16.f);
  }
}

// ---------------------------------------------------------------------------
// Gate tail: LayerNorm + exact GELU + Wg2 dot + combined softmax -> attn weights
// ---------------------------------------------------------------------------
__global__ __launch_bounds__(128) void gate_tail_kernel(
    const float* __restrict__ hbuf, const float* __restrict__ gamma,
    const float* __restrict__ beta, const float* __restrict__ Wg2,
    const float* __restrict__ bg2, const float* __restrict__ si,
    float* __restrict__ aw)
{
  const int  t = threadIdx.x;
  const long b = blockIdx.x;
  __shared__ float red[128];
  __shared__ float gl[5];
  const float* hb = hbuf + b * 512;

  float x[4];
#pragma unroll
  for (int i = 0; i < 4; ++i) x[i] = hb[t * 4 + i];

  red[t] = x[0] + x[1] + x[2] + x[3];
  __syncthreads();
  for (int off = 64; off > 0; off >>= 1) { if (t < off) red[t] += red[t + off]; __syncthreads(); }
  const float mean = red[0] * (1.f / 512.f);
  __syncthreads();

  float lv = 0.f;
#pragma unroll
  for (int i = 0; i < 4; ++i) { const float d = x[i] - mean; lv += d * d; }
  red[t] = lv;
  __syncthreads();
  for (int off = 64; off > 0; off >>= 1) { if (t < off) red[t] += red[t + off]; __syncthreads(); }
  const float rstd = rsqrtf(red[0] * (1.f / 512.f) + 1e-5f);
  __syncthreads();

  float xn[4];
#pragma unroll
  for (int i = 0; i < 4; ++i) {
    const int idx = t * 4 + i;
    const float v = (x[i] - mean) * rstd * gamma[idx] + beta[idx];
    xn[i] = 0.5f * v * (1.f + erff(v * 0.70710678118654752f));  // exact GELU
  }

  for (int s = 0; s < 5; ++s) {
    float p = 0.f;
#pragma unroll
    for (int i = 0; i < 4; ++i) p += xn[i] * Wg2[s * 512 + t * 4 + i];
    red[t] = p;
    __syncthreads();
    for (int off = 64; off > 0; off >>= 1) { if (t < off) red[t] += red[t + off]; __syncthreads(); }
    if (t == 0) gl[s] = red[0] + bg2[s];
    __syncthreads();
  }

  if (t == 0) {
    float cs[5]; float mx = -1e30f;
    for (int s = 0; s < 5; ++s) {
      cs[s] = 0.7f * si[b * 5 + s] + 0.3f * gl[s];
      mx = fmaxf(mx, cs[s]);
    }
    float den = 0.f;
    for (int s = 0; s < 5; ++s) { cs[s] = expf(cs[s] - mx); den += cs[s]; }
    const float inv = 1.f / den;
    for (int s = 0; s < 5; ++s) aw[b * 5 + s] = cs[s] * inv;
  }
}

// ---------------------------------------------------------------------------
// combined_logits[b,c] = sum_s aw[b,s] * logits_s[b,c]    (float4 over C=1000)
// ---------------------------------------------------------------------------
__global__ __launch_bounds__(256) void combine_logits_kernel(
    const float* __restrict__ l0, const float* __restrict__ l1,
    const float* __restrict__ l2, const float* __restrict__ l3,
    const float* __restrict__ l4, const float* __restrict__ aw,
    float* __restrict__ out)
{
  const long idx = (long)blockIdx.x * 256 + threadIdx.x;  // B * 250
  const long b  = idx / 250;
  const int  c4 = (int)(idx % 250);
  const float a0 = aw[b * 5 + 0], a1 = aw[b * 5 + 1], a2 = aw[b * 5 + 2],
              a3 = aw[b * 5 + 3], a4 = aw[b * 5 + 4];
  const long off = b * 1000 + (long)c4 * 4;
  const float4 r0 = *(const float4*)(l0 + off);
  const float4 r1 = *(const float4*)(l1 + off);
  const float4 r2 = *(const float4*)(l2 + off);
  const float4 r3 = *(const float4*)(l3 + off);
  const float4 r4 = *(const float4*)(l4 + off);
  float4 r;
  r.x = a0 * r0.x + a1 * r1.x + a2 * r2.x + a3 * r3.x + a4 * r4.x;
  r.y = a0 * r0.y + a1 * r1.y + a2 * r2.y + a3 * r3.y + a4 * r4.y;
  r.z = a0 * r0.z + a1 * r1.z + a2 * r2.z + a3 * r3.z + a4 * r4.z;
  r.w = a0 * r0.w + a1 * r1.w + a2 * r2.w + a3 * r3.w + a4 * r4.w;
  *(float4*)(out + off) = r;
}

// ---------------------------------------------------------------------------
// Launch
// ---------------------------------------------------------------------------
static inline void cvt_launch(const float* src, unsigned short* dst, long n,
                              hipStream_t stream) {
  const long n4 = n / 4;
  const long blocks = (n4 + 255) / 256;
  cvt_f32_to_bf16_kernel<<<dim3((unsigned)blocks), dim3(256), 0, stream>>>(src, dst, n4);
}

extern "C" void kernel_launch(void* const* d_in, const int* in_sizes, int n_in,
                              void* d_out, int out_size, void* d_ws, size_t ws_size,
                              hipStream_t stream) {
  (void)in_sizes; (void)n_in; (void)out_size; (void)ws_size;

  const float* features  = (const float*)d_in[0];
  const float* sf[5]     = {(const float*)d_in[1], (const float*)d_in[2],
                            (const float*)d_in[3], (const float*)d_in[4],
                            (const float*)d_in[5]};
  const float* logits[5] = {(const float*)d_in[6], (const float*)d_in[7],
                            (const float*)d_in[8], (const float*)d_in[9],
                            (const float*)d_in[10]};
  const float* Ws[5]     = {(const float*)d_in[11], (const float*)d_in[12],
                            (const float*)d_in[13], (const float*)d_in[14],
                            (const float*)d_in[15]};
  const float* bs        = (const float*)d_in[16];
  const float* scale_emb = (const float*)d_in[17];
  const float* Wq        = (const float*)d_in[18];
  const float* bq        = (const float*)d_in[19];
  const float* Wk        = (const float*)d_in[20];
  const float* bk        = (const float*)d_in[21];
  const float* Wv        = (const float*)d_in[22];
  const float* bv        = (const float*)d_in[23];
  const float* Wg1       = (const float*)d_in[24];
  const float* bg1       = (const float*)d_in[25];
  const float* g_gamma   = (const float*)d_in[26];
  const float* g_beta    = (const float*)d_in[27];
  const float* Wg2       = (const float*)d_in[28];
  const float* bg2       = (const float*)d_in[29];
  const float* temp      = (const float*)d_in[30];
  const int*   routes    = (const int*)d_in[31];

  const long scales[5] = {64, 128, 256, 512, 1024};

  // ---- workspace layout: bf16 region first, then f32 region ----
  unsigned short* u = (unsigned short*)d_ws;
  unsigned short* fbf      = u; u += 8388608L;                 // features bf16
  unsigned short* sfbf[5];
  for (int s = 0; s < 5; ++s) { sfbf[s] = u; u += 8192L * scales[s]; }
  unsigned short* wsbf[5];
  for (int s = 0; s < 5; ++s) { wsbf[s] = u; u += 1024L * scales[s]; }
  unsigned short* wqbf     = u; u += 1048576L;
  unsigned short* wkbf     = u; u += 1048576L;
  unsigned short* wvbf     = u; u += 1048576L;
  unsigned short* wg1bf    = u; u += 524288L;
  unsigned short* commonbf = u; u += 41943040L;                // (B*S, FD) bf16

  float* f = (float*)u;                                        // 16B-aligned
  float* Qbuf  = f; f += 8388608L;
  float* Kbuf  = f; f += 41943040L;
  float* Vbuf  = f; f += 41943040L;
  float* hbuf  = f; f += 4194304L;
  float* sibuf = f; f += 40960L;

  float* out = (float*)d_out;
  float* aw  = out + 8192L * 1000L;          // attention_weights tail of d_out

  // ---- 1) materialize bf16 operands ----
  cvt_launch(features, fbf, 8388608L, stream);
  for (int s = 0; s < 5; ++s) cvt_launch(sf[s], sfbf[s], 8192L * scales[s], stream);
  for (int s = 0; s < 5; ++s) cvt_launch(Ws[s], wsbf[s], 1024L * scales[s], stream);
  cvt_launch(Wq,  wqbf,  1048576L, stream);
  cvt_launch(Wk,  wkbf,  1048576L, stream);
  cvt_launch(Wv,  wvbf,  1048576L, stream);
  cvt_launch(Wg1, wg1bf, 524288L,  stream);

  const dim3 blk(256);

  // ---- 2) common[b, s, :] = sf_s @ Ws_s^T + bs[s] + scale_emb[s]  (bf16 out) ----
  for (int s = 0; s < 5; ++s) {
    gemm_nt_bf16_wmma<true, true><<<dim3(1024 / BN, 8192 / BM), blk, 0, stream>>>(
        sfbf[s], wsbf[s], bs + s * 1024, scale_emb + s * 1024,
        commonbf + s * 1024, (int)scales[s], 5120);
  }
  // ---- 3) projections (f32 out) ----
  gemm_nt_bf16_wmma<false, false><<<dim3(1024 / BN, 8192 / BM), blk, 0, stream>>>(
      fbf, wqbf, bq, nullptr, Qbuf, 1024, 1024);
  gemm_nt_bf16_wmma<false, false><<<dim3(1024 / BN, 40960 / BM), blk, 0, stream>>>(
      commonbf, wkbf, bk, nullptr, Kbuf, 1024, 1024);
  gemm_nt_bf16_wmma<false, false><<<dim3(1024 / BN, 40960 / BM), blk, 0, stream>>>(
      commonbf, wvbf, bv, nullptr, Vbuf, 1024, 1024);
  gemm_nt_bf16_wmma<false, false><<<dim3(512 / BN, 8192 / BM), blk, 0, stream>>>(
      fbf, wg1bf, bg1, nullptr, hbuf, 1024, 512);

  // ---- 4) routed attention -> scale_importance ----
  attn_route_kernel<<<dim3(8192 / 16), dim3(16, 16), 0, stream>>>(
      Qbuf, Kbuf, Vbuf, routes, temp, sibuf);

  // ---- 5) gate tail -> attention_weights (d_out tail) ----
  gate_tail_kernel<<<dim3(8192), dim3(128), 0, stream>>>(
      hbuf, g_gamma, g_beta, Wg2, bg2, sibuf, aw);

  // ---- 6) combined logits ----
  combine_logits_kernel<<<dim3((8192L * 250L) / 256L), blk, 0, stream>>>(
      logits[0], logits[1], logits[2], logits[3], logits[4], aw, out);
}
